// ContrastiveLoss_11166914970200
// MI455X (gfx1250) — matrix operand order repaired and checked by reading, not compiled, not run on hardware
//
#include <hip/hip_runtime.h>

// Problem constants (from setup_inputs): N=2048, d=256, T=256, Q=64
#define N_X   2048
#define DIMD  256
#define T_TR  256
#define Q_Q   64
#define TQ    (T_TR * Q_Q)        // 16384
#define INV_TEMP (1.0f / 0.3f)

typedef __attribute__((ext_vector_type(16))) __bf16 v16bf;
typedef __attribute__((ext_vector_type(8)))  float  v8f;

union Frag { int4 q[2]; v16bf v; };

__device__ __forceinline__ unsigned short f32_to_bf16_rne(float f) {
    unsigned int u = __float_as_uint(f);
    u += 0x7fffu + ((u >> 16) & 1u);
    return (unsigned short)(u >> 16);
}

// ---- split f32 -> bf16 hi + bf16 lo (error-compensated operands) -----------
__global__ void split_bf16_kernel(const float* __restrict__ in,
                                  unsigned short* __restrict__ hi,
                                  unsigned short* __restrict__ lo, int n) {
    int i = blockIdx.x * blockDim.x + threadIdx.x;
    if (i >= n) return;
    float f = in[i];
    unsigned short h = f32_to_bf16_rne(f);
    float fh = __uint_as_float(((unsigned int)h) << 16);
    hi[i] = h;
    lo[i] = f32_to_bf16_rne(f - fh);
}

__global__ void zero_kernel(float* __restrict__ p, int n) {
    int i = blockIdx.x * blockDim.x + threadIdx.x;
    if (i < n) p[i] = 0.0f;
}

// ---- fragment loaders per CDNA5 ISA 7.12.2 layouts -------------------------
// A (16x32 bf16): lanes 0-15 row M=lane, K {0-7 | 16-23}; lanes 16-31 same M,
//                 K {8-15 | 24-31}.  rowp points at row start (+hf*8 applied).
__device__ __forceinline__ v16bf load_fragA(const unsigned short* rowp, int k0) {
    Frag f;
    f.q[0] = *(const int4*)(rowp + k0);        // rel K 0-7 / 8-15
    f.q[1] = *(const int4*)(rowp + k0 + 16);   // rel K 16-23 / 24-31
    return f.v;
}
// B (32x16 bf16): lane n (0-15) holds col n, K {0-7 | 16-23}; lanes 16-31 K {8-15 | 24-31}.
__device__ __forceinline__ v16bf load_fragB(const unsigned short* rowp, int k0, int hf) {
    Frag f;
    f.q[0] = *(const int4*)(rowp + k0 + hf * 8);
    f.q[1] = *(const int4*)(rowp + k0 + hf * 8 + 16);
    return f.v;
}

#define WMMA_BF16(A, B, C) \
    __builtin_amdgcn_wmma_f32_16x16x32_bf16(false, (A), false, (B), (short)0, (C), false, false)

// ---- S_xy = exp(x @ yf^T / T): per-row total + positive(track) sums --------
__global__ __launch_bounds__(32)
void xy_kernel(const unsigned short* __restrict__ xh, const unsigned short* __restrict__ xl,
               const unsigned short* __restrict__ yh, const unsigned short* __restrict__ yl,
               const int* __restrict__ tid,
               float* __restrict__ rxy_total, float* __restrict__ rxy_pos) {
    const int lane = threadIdx.x;
    const int hf   = lane >> 4;      // which half of the wave
    const int mrow = lane & 15;
    const int i0     = blockIdx.x * 16;    // row tile
    const int j0base = blockIdx.y * 256;   // 16 column tiles per wave

    int tids8[8];
#pragma unroll
    for (int v = 0; v < 8; ++v) tids8[v] = tid[i0 + hf * 8 + v];

    const unsigned short* arh = xh + (size_t)(i0 + mrow) * DIMD + hf * 8;
    const unsigned short* arl = xl + (size_t)(i0 + mrow) * DIMD + hf * 8;

    float acc_tot[8], acc_pos[8];
#pragma unroll
    for (int v = 0; v < 8; ++v) { acc_tot[v] = 0.f; acc_pos[v] = 0.f; }

    for (int jt = 0; jt < 16; ++jt) {
        const int j0 = j0base + jt * 16;
        const unsigned short* brh = yh + (size_t)(j0 + mrow) * DIMD;
        const unsigned short* brl = yl + (size_t)(j0 + mrow) * DIMD;
        v8f c = {0.f, 0.f, 0.f, 0.f, 0.f, 0.f, 0.f, 0.f};
#pragma unroll
        for (int kt = 0; kt < 8; ++kt) {
            const int k0 = kt * 32;
            v16bf ah = load_fragA(arh, k0);
            v16bf al = load_fragA(arl, k0);
            v16bf bh = load_fragB(brh, k0, hf);
            v16bf bl = load_fragB(brl, k0, hf);
            c = WMMA_BF16(ah, bh, c);
            c = WMMA_BF16(ah, bl, c);
            c = WMMA_BF16(al, bh, c);
        }
        const int colmod = (j0 + mrow) & (T_TR - 1);   // buggy-but-faithful j % T
#pragma unroll
        for (int v = 0; v < 8; ++v) {
            float e = __expf(c[v] * INV_TEMP);
            acc_tot[v] += e;
            if (colmod == tids8[v]) acc_pos[v] += e;
        }
    }
    // reduce across each 16-lane half (C/D layout: rows v / v+8 per half)
#pragma unroll
    for (int v = 0; v < 8; ++v) {
#pragma unroll
        for (int m = 1; m < 16; m <<= 1) {
            acc_tot[v] += __shfl_xor(acc_tot[v], m);
            acc_pos[v] += __shfl_xor(acc_pos[v], m);
        }
    }
    if (mrow == 0) {
#pragma unroll
        for (int v = 0; v < 8; ++v) {
            int row = i0 + hf * 8 + v;
            atomicAdd(&rxy_total[row], acc_tot[v]);
            atomicAdd(&rxy_pos[row],  acc_pos[v]);
        }
    }
}

// ---- S_xx = exp(x @ x^T / T): per-row total + same-track sums + diagonal ---
__global__ __launch_bounds__(32)
void xx_kernel(const unsigned short* __restrict__ xh, const unsigned short* __restrict__ xl,
               const int* __restrict__ tid,
               float* __restrict__ rxx_total, float* __restrict__ rxx_pos,
               float* __restrict__ diag) {
    const int lane = threadIdx.x;
    const int hf   = lane >> 4;
    const int mrow = lane & 15;
    const int i0     = blockIdx.x * 16;
    const int j0base = blockIdx.y * 256;

    int tids8[8];
#pragma unroll
    for (int v = 0; v < 8; ++v) tids8[v] = tid[i0 + hf * 8 + v];

    const unsigned short* arh = xh + (size_t)(i0 + mrow) * DIMD + hf * 8;
    const unsigned short* arl = xl + (size_t)(i0 + mrow) * DIMD + hf * 8;

    float acc_tot[8], acc_pos[8];
#pragma unroll
    for (int v = 0; v < 8; ++v) { acc_tot[v] = 0.f; acc_pos[v] = 0.f; }

    for (int jt = 0; jt < 16; ++jt) {
        const int j0 = j0base + jt * 16;
        const int tcol = tid[j0 + mrow];
        const unsigned short* brh = xh + (size_t)(j0 + mrow) * DIMD;
        const unsigned short* brl = xl + (size_t)(j0 + mrow) * DIMD;
        v8f c = {0.f, 0.f, 0.f, 0.f, 0.f, 0.f, 0.f, 0.f};
#pragma unroll
        for (int kt = 0; kt < 8; ++kt) {
            const int k0 = kt * 32;
            v16bf ah = load_fragA(arh, k0);
            v16bf al = load_fragA(arl, k0);
            v16bf bh = load_fragB(brh, k0, hf);
            v16bf bl = load_fragB(brl, k0, hf);
            c = WMMA_BF16(ah, bh, c);
            c = WMMA_BF16(ah, bl, c);
            c = WMMA_BF16(al, bh, c);
        }
        const int col = j0 + mrow;
#pragma unroll
        for (int v = 0; v < 8; ++v) {
            float e = __expf(c[v] * INV_TEMP);
            acc_tot[v] += e;
            if (tcol == tids8[v]) acc_pos[v] += e;
            int row = i0 + hf * 8 + v;
            if (col == row) diag[row] = e;        // unique writer per n
        }
    }
#pragma unroll
    for (int v = 0; v < 8; ++v) {
#pragma unroll
        for (int m = 1; m < 16; m <<= 1) {
            acc_tot[v] += __shfl_xor(acc_tot[v], m);
            acc_pos[v] += __shfl_xor(acc_pos[v], m);
        }
    }
    if (mrow == 0) {
#pragma unroll
        for (int v = 0; v < 8; ++v) {
            int row = i0 + hf * 8 + v;
            atomicAdd(&rxx_total[row], acc_tot[v]);
            atomicAdd(&rxx_pos[row],  acc_pos[v]);
        }
    }
}

// ---- per-track segment sum + masked-mean log loss --------------------------
__global__ __launch_bounds__(T_TR)
void finalize_kernel(const float* __restrict__ rxy_total, const float* __restrict__ rxy_pos,
                     const float* __restrict__ rxx_total, const float* __restrict__ rxx_pos,
                     const float* __restrict__ diag, const int* __restrict__ tid,
                     float* __restrict__ out) {
    __shared__ float num_s[T_TR];
    __shared__ float den_s[T_TR];
    __shared__ int   cnt_s[T_TR];
    const int t = threadIdx.x;
    num_s[t] = 0.f; den_s[t] = 0.f; cnt_s[t] = 0;
    __syncthreads();
    for (int n = t; n < N_X; n += T_TR) {
        int   tn  = tid[n];
        float rp  = rxy_pos[n];
        float xpf = rxx_pos[n];
        float num_i = rp + 0.5f * (xpf - diag[n]);
        float den_i = (rxy_total[n] - rp) + (rxx_total[n] - xpf);
        atomicAdd(&num_s[tn], num_i);
        atomicAdd(&den_s[tn], den_i);
        atomicAdd(&cnt_s[tn], 1);
    }
    __syncthreads();
    float loss = 0.f, pres = 0.f;
    if (cnt_s[t] > 0) {
        float nu = num_s[t], de = den_s[t];
        loss = -__logf(nu / (de + nu));
        pres = 1.f;
    }
    __syncthreads();
    num_s[t] = loss; den_s[t] = pres;
    __syncthreads();
    for (int s = T_TR / 2; s > 0; s >>= 1) {
        if (t < s) { num_s[t] += num_s[t + s]; den_s[t] += den_s[t + s]; }
        __syncthreads();
    }
    if (t == 0) out[0] = num_s[0] / den_s[0];
}

extern "C" void kernel_launch(void* const* d_in, const int* in_sizes, int n_in,
                              void* d_out, int out_size, void* d_ws, size_t ws_size,
                              hipStream_t stream) {
    const float* x   = (const float*)d_in[0];
    const int*   tid = (const int*)d_in[1];
    const float* y   = (const float*)d_in[2];
    float* out = (float*)d_out;

    char* ws = (char*)d_ws;
    size_t off = 0;
    unsigned short* xh = (unsigned short*)(ws + off); off += (size_t)N_X * DIMD * 2;
    unsigned short* xl = (unsigned short*)(ws + off); off += (size_t)N_X * DIMD * 2;
    unsigned short* yh = (unsigned short*)(ws + off); off += (size_t)TQ  * DIMD * 2;
    unsigned short* yl = (unsigned short*)(ws + off); off += (size_t)TQ  * DIMD * 2;
    float* rxy_total = (float*)(ws + off); off += (size_t)N_X * 4;
    float* rxy_pos   = (float*)(ws + off); off += (size_t)N_X * 4;
    float* rxx_total = (float*)(ws + off); off += (size_t)N_X * 4;
    float* rxx_pos   = (float*)(ws + off); off += (size_t)N_X * 4;
    float* diag      = (float*)(ws + off); off += (size_t)N_X * 4;

    // 1) convert operands to split-bf16
    {
        int nx = N_X * DIMD;
        split_bf16_kernel<<<(nx + 255) / 256, 256, 0, stream>>>(x, xh, xl, nx);
        int ny = TQ * DIMD;
        split_bf16_kernel<<<(ny + 255) / 256, 256, 0, stream>>>(y, yh, yl, ny);
    }
    // 2) zero accumulators (rxy_total..rxx_pos are contiguous; diag is fully overwritten)
    zero_kernel<<<(4 * N_X + 255) / 256, 256, 0, stream>>>(rxy_total, 4 * N_X);

    // 3) fused GEMM + exp + row reductions
    xy_kernel<<<dim3(N_X / 16, TQ / 256), 32, 0, stream>>>(xh, xl, yh, yl, tid,
                                                           rxy_total, rxy_pos);
    xx_kernel<<<dim3(N_X / 16, N_X / 256), 32, 0, stream>>>(xh, xl, tid,
                                                            rxx_total, rxx_pos, diag);

    // 4) per-track segment sums + scalar loss
    finalize_kernel<<<1, T_TR, 0, stream>>>(rxy_total, rxy_pos, rxx_total, rxx_pos,
                                            diag, tid, out);
}
